// TurboQuantWrapper_5428838662722
// MI455X (gfx1250) — matrix operand order repaired and checked by reading, not compiled, not run on hardware
//
#include <hip/hip_runtime.h>
#include <hip/hip_bf16.h>

typedef _Float16 v16h __attribute__((ext_vector_type(16)));
typedef _Float16 v8h  __attribute__((ext_vector_type(8)));
typedef _Float16 v4h  __attribute__((ext_vector_type(4)));
typedef float    v8f  __attribute__((ext_vector_type(8)));
typedef int      v4i  __attribute__((ext_vector_type(4)));

#define AS1 __attribute__((address_space(1)))
#define AS3 __attribute__((address_space(3)))

#define IN_F     4096
#define OUT_F    11008
#define M_ROWS   4096
#define GROUPSZ  128
#define N_GROUPS 32
#define LDT      40   // LDS row stride in halves (80B: 16B-aligned, bank-spread)

#if (__has_builtin(__builtin_amdgcn_global_load_async_to_lds_b128) && \
     __has_builtin(__builtin_amdgcn_s_wait_asynccnt))
#define HAS_ASYNC_LDS 1
#else
#define HAS_ASYNC_LDS 0
#endif

// ---------------------------------------------------------------------------
// Kernel 1: dequantize 4-bit codes -> f16 weights (one wave per 128-elem group)
// ---------------------------------------------------------------------------
__global__ __launch_bounds__(256) void dequant_kernel(
    const int*   __restrict__ packed,
    const float* __restrict__ norms,
    const float* __restrict__ centroids,
    const float* __restrict__ signs1,
    const float* __restrict__ signs2,
    _Float16*    __restrict__ w16)
{
    const int lane = threadIdx.x & 31;
    const int wave = threadIdx.x >> 5;
    const long gid = (long)blockIdx.x * 8 + wave;   // group id in [0, OUT_F*N_GROUPS)
    const int  o   = (int)(gid >> 5);               // output feature
    const int  g   = (int)(gid & 31);               // group within row

    // --- decode: lane holds elements 4l..4l+3 (bytes 2l, 2l+1: lo then hi) ---
    const int* pw = packed + (size_t)o * (IN_F / 2) + g * 64 + 2 * lane;
    const int p0 = pw[0];
    const int p1 = pw[1];

    const float cv = centroids[lane & 15];          // LUT broadcast across lanes
    float v0 = __shfl(cv,  p0        & 15);
    float v1 = __shfl(cv, (p0 >> 4)  & 15);
    float v2 = __shfl(cv,  p1        & 15);
    float v3 = __shfl(cv, (p1 >> 4)  & 15);

    const float4 s2 = *((const float4*)signs2 + lane);
    const float4 s1 = *((const float4*)signs1 + lane);

    // --- FWHT 128 (unnormalized), x2, with sign flips between ---
    #pragma unroll
    for (int pass = 0; pass < 2; ++pass) {
        float a, b;
        a = v0; b = v1; v0 = a + b; v1 = a - b;     // h = 1
        a = v2; b = v3; v2 = a + b; v3 = a - b;
        a = v0; b = v2; v0 = a + b; v2 = a - b;     // h = 2
        a = v1; b = v3; v1 = a + b; v3 = a - b;
        #pragma unroll
        for (int m = 1; m <= 16; m <<= 1) {         // h = 4..64 via lane xor
            const bool hi = (lane & m) != 0;
            const float o0 = __shfl_xor(v0, m);
            const float o1 = __shfl_xor(v1, m);
            const float o2 = __shfl_xor(v2, m);
            const float o3 = __shfl_xor(v3, m);
            v0 = hi ? (o0 - v0) : (v0 + o0);
            v1 = hi ? (o1 - v1) : (v1 + o1);
            v2 = hi ? (o2 - v2) : (v2 + o2);
            v3 = hi ? (o3 - v3) : (v3 + o3);
        }
        if (pass == 0) { v0 *= s2.x; v1 *= s2.y; v2 *= s2.z; v3 *= s2.w; }
        else           { v0 *= s1.x; v1 *= s1.y; v2 *= s1.z; v3 *= s1.w; }
    }

    // fold both 1/sqrt(128) orthonormal scales into the norm
    const float nrm = norms[(size_t)o * N_GROUPS + g] * (1.0f / 128.0f);
    v4h w4;
    w4.x = (_Float16)(v0 * nrm);
    w4.y = (_Float16)(v1 * nrm);
    w4.z = (_Float16)(v2 * nrm);
    w4.w = (_Float16)(v3 * nrm);
    *(v4h*)(w16 + (size_t)o * IN_F + g * GROUPSZ + 4 * lane) = w4;
}

// ---------------------------------------------------------------------------
// Kernel 2: x (f32) -> f16
// ---------------------------------------------------------------------------
__global__ __launch_bounds__(256) void x2h_kernel(
    const float* __restrict__ x, _Float16* __restrict__ xh)
{
    const size_t i = ((size_t)blockIdx.x * 256 + threadIdx.x) * 4;
    const float4 f = *(const float4*)(x + i);
    v4h h4;
    h4.x = (_Float16)f.x; h4.y = (_Float16)f.y;
    h4.z = (_Float16)f.z; h4.w = (_Float16)f.w;
    *(v4h*)(xh + i) = h4;
}

// ---------------------------------------------------------------------------
// GEMM helpers
// ---------------------------------------------------------------------------
union FragU { v16h v; v8h h[2]; };

#if HAS_ASYNC_LDS
// One async 16B global->LDS copy (no VGPR landing, tracked by ASYNCcnt).
__device__ __forceinline__ void async_cp16(const _Float16* g, _Float16* l)
{
    // generic -> AS1/AS3 addrspacecasts first, then retype to int4 pointers.
    const AS1 _Float16* g1 = (const AS1 _Float16*)g;
    AS3 _Float16*       l3 = (AS3 _Float16*)l;
    __builtin_amdgcn_global_load_async_to_lds_b128(
        (AS1 v4i*)g1, (AS3 v4i*)l3, /*offset=*/0, /*cpol=*/0);
}

// Issue 4 async 16B copies per thread: one 128x32-half tile of A and of B.
__device__ __forceinline__ void stage_async(
    const _Float16* __restrict__ gA, const _Float16* __restrict__ gB,
    _Float16* sA, _Float16* sB, int tid)
{
    #pragma unroll
    for (int i = 0; i < 2; ++i) {
        const int e   = i * 2048 + tid * 8;
        const int row = e >> 5;
        const int col = e & 31;
        async_cp16(gA + (size_t)row * IN_F + col, sA + row * LDT + col);
        async_cp16(gB + (size_t)row * IN_F + col, sB + row * LDT + col);
    }
}
#endif

// One 32-deep K step: 4 A frags x 2 B frags -> 8 WMMA.
__device__ __forceinline__ void compute_step(
    const _Float16* sA, const _Float16* sB,
    v8f (&acc)[4][2], int wm, int wn, int lr, int hl)
{
    v16h bfrag[2];
    #pragma unroll
    for (int u = 0; u < 2; ++u) {
        FragU f;
        const _Float16* p = sB + (wn + 16 * u + lr) * LDT + hl * 8;
        f.h[0] = *(const v8h*)(p);
        f.h[1] = *(const v8h*)(p + 16);
        bfrag[u] = f.v;
    }
    #pragma unroll
    for (int t = 0; t < 4; ++t) {
        FragU f;
        const _Float16* p = sA + (wm + 16 * t + lr) * LDT + hl * 8;
        f.h[0] = *(const v8h*)(p);
        f.h[1] = *(const v8h*)(p + 16);
        const v16h afrag = f.v;
        #pragma unroll
        for (int u = 0; u < 2; ++u) {
            acc[t][u] = __builtin_amdgcn_wmma_f32_16x16x32_f16(
                false, afrag, false, bfrag[u], (short)0, acc[t][u], false, false);
        }
    }
}

// ---------------------------------------------------------------------------
// Kernel 3: tiled WMMA GEMM  y[m][n] = sum_k xh[m][k]*wh[n][k] + bias[n]
//   block: 256 threads (8 waves), 128x128 output tile, K step 32
//   wave: 64(M) x 32(N) sub-tile = 4x2 v_wmma_f32_16x16x32_f16 accumulators
//   double-buffered LDS with async-to-LDS prefetch when available
// ---------------------------------------------------------------------------
__global__ __launch_bounds__(256) void gemm_kernel(
    const _Float16* __restrict__ xh,
    const _Float16* __restrict__ wh,
    const float*    __restrict__ bias,
    float*          __restrict__ out)
{
    __shared__ _Float16 sA[2][128 * LDT];
    __shared__ _Float16 sB[2][128 * LDT];

    const int tid  = threadIdx.x;
    const int lane = tid & 31;
    const int wave = tid >> 5;
    const int lr   = lane & 15;     // row within 16-lane half
    const int hl   = lane >> 4;     // which half-wave

    const int nBase = blockIdx.x * 128;
    const int mBase = blockIdx.y * 128;
    const int wm    = (wave & 1) * 64;
    const int wn    = (wave >> 1) * 32;

    const _Float16* gA = xh + (size_t)mBase * IN_F;
    const _Float16* gB = wh + (size_t)nBase * IN_F;

    v8f acc[4][2] = {};

#if HAS_ASYNC_LDS
    const int NK = IN_F / 32;
    stage_async(gA, gB, sA[0], sB[0], tid);           // prime buffer 0
    for (int kk = 0; kk < NK - 1; ++kk) {
        const int cur = kk & 1;
        // prefetch next tile into the other buffer (overlaps with compute)
        stage_async(gA + (kk + 1) * 32, gB + (kk + 1) * 32,
                    sA[cur ^ 1], sB[cur ^ 1], tid);
        __builtin_amdgcn_s_wait_asynccnt(4);          // prev batch (4 ops) done
        __syncthreads();
        compute_step(sA[cur], sB[cur], acc, wm, wn, lr, hl);
        __syncthreads();                              // cur safe to overwrite at kk+2
    }
    __builtin_amdgcn_s_wait_asynccnt(0);              // tail tile complete
    __syncthreads();
    compute_step(sA[(NK - 1) & 1], sB[(NK - 1) & 1], acc, wm, wn, lr, hl);
#else
    for (int k0 = 0; k0 < IN_F; k0 += 32) {
        __syncthreads();
        #pragma unroll
        for (int i = 0; i < 2; ++i) {
            const int e   = i * 2048 + tid * 8;
            const int row = e >> 5;
            const int col = e & 31;
            *(v8h*)(sA[0] + row * LDT + col) =
                *(const v8h*)(gA + (size_t)row * IN_F + k0 + col);
            *(v8h*)(sB[0] + row * LDT + col) =
                *(const v8h*)(gB + (size_t)row * IN_F + k0 + col);
        }
        __syncthreads();
        compute_step(sA[0], sB[0], acc, wm, wn, lr, hl);
    }
#endif

    // epilogue: C/D layout: VGPR i, lane -> row = 8*hl + i, col = lr
    #pragma unroll
    for (int u = 0; u < 2; ++u) {
        const int col = nBase + wn + 16 * u + lr;
        const float bv = bias[col];
        #pragma unroll
        for (int t = 0; t < 4; ++t) {
            const int row0 = mBase + wm + 16 * t + 8 * hl;
            float* op = out + (size_t)row0 * OUT_F + col;
            #pragma unroll
            for (int i = 0; i < 8; ++i) {
                op[(size_t)i * OUT_F] = acc[t][u][i] + bv;
            }
        }
    }
}

// ---------------------------------------------------------------------------
extern "C" void kernel_launch(void* const* d_in, const int* in_sizes, int n_in,
                              void* d_out, int out_size, void* d_ws, size_t ws_size,
                              hipStream_t stream)
{
    const float* x         = (const float*)d_in[0];
    const int*   packed    = (const int*)  d_in[1];
    const float* norms     = (const float*)d_in[2];
    const float* centroids = (const float*)d_in[3];
    const float* signs1    = (const float*)d_in[4];
    const float* signs2    = (const float*)d_in[5];
    const float* bias      = (const float*)d_in[6];
    float* out = (float*)d_out;

    _Float16* wh = (_Float16*)d_ws;                    // OUT_F * IN_F halves (~90 MB)
    _Float16* xh = wh + (size_t)OUT_F * IN_F;          // M_ROWS * IN_F halves (~34 MB)

    // 1) dequantize weights to f16 (one wave per group; 8 waves per block)
    dequant_kernel<<<(OUT_F * N_GROUPS) / 8, 256, 0, stream>>>(
        packed, norms, centroids, signs1, signs2, wh);

    // 2) convert activations to f16
    x2h_kernel<<<(M_ROWS * (size_t)IN_F) / (4 * 256), 256, 0, stream>>>(x, xh);

    // 3) WMMA GEMM + bias
    gemm_kernel<<<dim3(OUT_F / 128, M_ROWS / 128), 256, 0, stream>>>(
        xh, wh, bias, out);
}
